// TransformerCrossAttBlock_60928406061442
// MI455X (gfx1250) — matrix operand order repaired and checked by reading, not compiled, not run on hardware
//
#include <hip/hip_runtime.h>
#include <hip/hip_bf16.h>

// ---------------------------------------------------------------------------
// Problem constants (fixed by the reference)
// ---------------------------------------------------------------------------
constexpr int Bc = 2, SQc = 2048, SKVc = 2048, Dc = 256, Hc = 8, MLPc = 1024;
constexpr int HDc = Dc / Hc; // 32

typedef __attribute__((ext_vector_type(16))) _Float16 v16h;
typedef __attribute__((ext_vector_type(8)))  _Float16 v8h;
typedef __attribute__((ext_vector_type(8)))  float    v8f;

// v_wmma_f32_16x16x32_f16 : D = A(16x32 f16) * B(32x16 f16) + C(16x16 f32)
__device__ inline v8f wmma_f16(v16h a, v16h b, v8f c) {
  return __builtin_amdgcn_wmma_f32_16x16x32_f16(
      /*neg_a=*/false, a, /*neg_b=*/false, b,
      /*c_mod=*/(short)0, c, /*reuse_a=*/false, /*reuse_b=*/false);
}

// Build a 16-half fragment from two contiguous 16-byte chunks.
__device__ inline v16h load16(const _Float16* p0, const _Float16* p1) {
  v8h lo = *(const v8h*)p0;
  v8h hi = *(const v8h*)p1;
  v16h r;
#pragma unroll
  for (int i = 0; i < 8; ++i) { r[i] = lo[i]; r[i + 8] = hi[i]; }
  return r;
}

// ---------------------------------------------------------------------------
// LayerNorm over rows of length D=256 -> f16 output. One block per row.
// ---------------------------------------------------------------------------
__global__ __launch_bounds__(256) void ln_rows(const float* __restrict__ x,
                                               const float* __restrict__ g,
                                               const float* __restrict__ b,
                                               _Float16* __restrict__ out) {
  __shared__ float red[256];
  const int r = blockIdx.x;
  const int t = threadIdx.x;
  const float v = x[(size_t)r * Dc + t];

  red[t] = v;
  __syncthreads();
  for (int s = 128; s > 0; s >>= 1) {
    if (t < s) red[t] += red[t + s];
    __syncthreads();
  }
  const float mu = red[0] * (1.0f / Dc);
  __syncthreads();

  const float d = v - mu;
  red[t] = d * d;
  __syncthreads();
  for (int s = 128; s > 0; s >>= 1) {
    if (t < s) red[t] += red[t + s];
    __syncthreads();
  }
  const float var = red[0] * (1.0f / Dc);
  const float y = d * rsqrtf(var + 1e-6f) * g[t] + b[t];
  out[(size_t)r * Dc + t] = (_Float16)y;
}

// ---------------------------------------------------------------------------
// Convert + transpose weights: W[K,N] f32 -> Wt[N,K] f16
// ---------------------------------------------------------------------------
__global__ __launch_bounds__(256) void cvt_t(const float* __restrict__ W,
                                             _Float16* __restrict__ Wt,
                                             int K, int N) {
  int i = blockIdx.x * 256 + threadIdx.x;
  if (i < K * N) {
    int k = i / N, n = i - k * N;
    Wt[(size_t)n * K + k] = (_Float16)W[i];
  }
}

// ---------------------------------------------------------------------------
// Generic WMMA GEMM: C[M,N] = A[M,K](f16,row-major) * Bt[N,K]^T + bias
// One 16x16 C tile per wave, K-loop in chunks of 32.
// Epilogue index: b=m/S, s=m%S, h=n/HDp, hd=n%HDp; idx=b*c0+s*c1+h*c2+hd*c3
// Optional exact GELU, optional f32 residual add (residual[m*N+n]).
// ---------------------------------------------------------------------------
__global__ __launch_bounds__(128) void gemm_wmma_kernel(
    const _Float16* __restrict__ A, const _Float16* __restrict__ Bt,
    const float* __restrict__ bias, const float* __restrict__ residual,
    float* __restrict__ outF32, _Float16* __restrict__ outF16,
    int M, int N, int K, int S, int HDp,
    long long c0, long long c1, long long c2, long long c3, int do_gelu) {
  const int wave = threadIdx.x >> 5;
  const int lane = threadIdx.x & 31;
  const int tile = blockIdx.x * (blockDim.x >> 5) + wave;
  const int ntn = N >> 4;
  const int tm = tile / ntn;
  const int tn = tile - tm * ntn;
  const int row = lane & 15;
  const int hiH = (lane < 16) ? 0 : 1;

  // A fragment: lane holds row (lane&15); K = kc + hiH*8 + {0..7, 16..23}
  const _Float16* arow = A + (size_t)(tm * 16 + row) * K + hiH * 8;
  // B fragment: lane holds col (lane&15); K = kc + hiH*8 + {0..7, 16..23}
  const _Float16* brow = Bt + (size_t)(tn * 16 + row) * K + hiH * 8;

  v8f acc = {};
  for (int kc = 0; kc < K; kc += 32) {
    __builtin_prefetch(arow + kc + 128, 0, 1); // global_prefetch_b8
    __builtin_prefetch(brow + kc + 128, 0, 1);
    v16h a = load16(arow + kc, arow + kc + 16);
    v16h b = load16(brow + kc, brow + kc + 16);
    acc = wmma_f16(a, b, acc);
  }

#pragma unroll
  for (int i = 0; i < 8; ++i) {
    const int m = tm * 16 + i + hiH * 8;
    const int n = tn * 16 + row;
    float v = acc[i] + (bias ? bias[n] : 0.0f);
    if (do_gelu) v = 0.5f * v * (1.0f + erff(v * 0.70710678118654752f));
    if (residual) v += residual[(size_t)m * N + n];
    const int bb = m / S;
    const int s = m - bb * S;
    const int h = n / HDp;
    const int hd = n - h * HDp;
    const size_t idx = (size_t)bb * c0 + (size_t)s * c1 + (size_t)h * c2 + (size_t)hd * c3;
    if (outF16) outF16[idx] = (_Float16)v;
    else        outF32[idx] = v;
  }
}

// ---------------------------------------------------------------------------
// Fused attention: per block = one (b,h) head, 16 query rows.
// 4 waves; the full 16 x 2048 f32 score strip lives in LDS (320KB/WGP budget).
//   Phase 1: scores = (Q K^T)/sqrt(HD) via WMMA, waves split SKV
//   Phase 2: softmax across the strip, write normalized weights to d_out
//   Phase 3: attn = W * V via WMMA (weights f32->f16 from LDS), cross-wave
//            partial reduction in LDS, wave 0 writes f16 attn for the O-proj.
// ---------------------------------------------------------------------------
__global__ __launch_bounds__(128) void attn_fused_kernel(
    const _Float16* __restrict__ qh,   // [B,H,SQ,HD]
    const _Float16* __restrict__ kh,   // [B,H,SKV,HD]
    const _Float16* __restrict__ vth,  // [B,H,HD,SKV]
    float* __restrict__ w_out,         // [B,H,SQ,SKV]
    _Float16* __restrict__ attn_h) {   // [B,SQ,D]
  constexpr int SKVP = SKVc + 4; // pad to dodge LDS bank conflicts
  extern __shared__ float smem[];
  float* sc    = smem;                    // [16][SKVP] score strip
  float* part  = sc + 16 * SKVP;          // [4][2][8][32] partial attn
  float* red   = part + 4 * 2 * 8 * 32;   // [16][8]
  float* rstat = red + 16 * 8;            // [16] rowmax then 1/rowsum

  const int tid  = threadIdx.x;
  const int wave = tid >> 5;
  const int lane = tid & 31;
  const int bh   = blockIdx.x / (SQc / 16);
  const int qb   = blockIdx.x % (SQc / 16);
  const int row  = lane & 15;
  const int hiH  = (lane < 16) ? 0 : 1;
  const float scale = 0.17677669529663687f; // 1/sqrt(HD=32)

  // ---- Phase 1: score strip ----
  const _Float16* qbase = qh + ((size_t)bh * SQc + qb * 16) * HDc;
  const v16h aq = load16(qbase + row * HDc + hiH * 8,
                         qbase + row * HDc + hiH * 8 + 16);
  for (int t = 0; t < SKVc / 16 / 4; ++t) {          // 32 tiles per wave
    const int kv0 = (wave * (SKVc / 16 / 4) + t) * 16;
    const _Float16* kp = kh + ((size_t)bh * SKVc + kv0 + row) * HDc + hiH * 8;
    v16h bkT = load16(kp, kp + 16);
    v8f c = {};
    c = wmma_f16(aq, bkT, c);
#pragma unroll
    for (int i = 0; i < 8; ++i)
      sc[(i + hiH * 8) * SKVP + kv0 + row] = c[i] * scale;
  }
  __syncthreads();

  // ---- Phase 2: softmax over each of the 16 rows ----
  {
    const int r = tid & 15, seg = tid >> 4; // 8 segments of 256 columns
    float m = -3.0e38f;
    for (int c0 = seg * 256; c0 < seg * 256 + 256; ++c0)
      m = fmaxf(m, sc[r * SKVP + c0]);
    red[r * 8 + seg] = m;
    __syncthreads();
    if (tid < 16) {
      float mm = red[tid * 8];
      for (int s = 1; s < 8; ++s) mm = fmaxf(mm, red[tid * 8 + s]);
      rstat[tid] = mm;
    }
    __syncthreads();
    const float rmax = rstat[r];
    float psum = 0.0f;
    for (int c0 = seg * 256; c0 < seg * 256 + 256; ++c0) {
      float e = expf(sc[r * SKVP + c0] - rmax);
      sc[r * SKVP + c0] = e;
      psum += e;
    }
    __syncthreads();
    red[r * 8 + seg] = psum;
    __syncthreads();
    if (tid < 16) {
      float s = 0.0f;
      for (int q = 0; q < 8; ++q) s += red[tid * 8 + q];
      rstat[tid] = 1.0f / s;
    }
    __syncthreads();
  }

  // normalize in LDS and stream weights to global (coalesced)
  float* wrow = w_out + ((size_t)bh * SQc + qb * 16) * SKVc;
  for (int e = tid; e < 16 * SKVc; e += 128) {
    const int r = e >> 11;          // SKV = 2048
    const int c = e & (SKVc - 1);
    const float v = sc[r * SKVP + c] * rstat[r];
    sc[r * SKVP + c] = v;
    wrow[(size_t)r * SKVc + c] = v;
  }
  __syncthreads();

  // ---- Phase 3: attn = W * V, waves split the SKV (K) dimension ----
  v8f acc0 = {}, acc1 = {};
  for (int t = 0; t < 16; ++t) {
    const int kc = wave * (SKVc / 4) + t * 32;
    const int kb2 = kc + hiH * 8;
    v16h aw;
#pragma unroll
    for (int j = 0; j < 8; ++j) {
      aw[j]     = (_Float16)sc[row * SKVP + kb2 + j];
      aw[j + 8] = (_Float16)sc[row * SKVP + kb2 + 16 + j];
    }
    const _Float16* vp0 = vth + ((size_t)bh * HDc + row) * SKVc + kc + hiH * 8;
    const _Float16* vp1 = vp0 + (size_t)16 * SKVc;
    v16h b0 = load16(vp0, vp0 + 16);
    v16h b1 = load16(vp1, vp1 + 16);
    acc0 = wmma_f16(aw, b0, acc0);
    acc1 = wmma_f16(aw, b1, acc1);
  }
#pragma unroll
  for (int i = 0; i < 8; ++i) {
    part[((wave * 2 + 0) * 8 + i) * 32 + lane] = acc0[i];
    part[((wave * 2 + 1) * 8 + i) * 32 + lane] = acc1[i];
  }
  __syncthreads();
  if (wave == 0) {
    const int b = bh / Hc, h = bh % Hc;
    for (int nt = 0; nt < 2; ++nt) {
#pragma unroll
      for (int i = 0; i < 8; ++i) {
        float s = 0.0f;
        for (int w = 0; w < 4; ++w) s += part[((w * 2 + nt) * 8 + i) * 32 + lane];
        const int m = i + hiH * 8;
        const int n = nt * 16 + row;
        attn_h[((size_t)b * SQc + qb * 16 + m) * Dc + h * HDc + n] = (_Float16)s;
      }
    }
  }
}

// ---------------------------------------------------------------------------
// Host-side orchestration
// ---------------------------------------------------------------------------
extern "C" void kernel_launch(void* const* d_in, const int* in_sizes, int n_in,
                              void* d_out, int out_size, void* d_ws, size_t ws_size,
                              hipStream_t stream) {
  (void)in_sizes; (void)n_in; (void)out_size; (void)ws_size;
  const float* qin  = (const float*)d_in[0];
  const float* kvin = (const float*)d_in[1];
  const float* ln1g = (const float*)d_in[2];
  const float* ln1b = (const float*)d_in[3];
  const float* ln2g = (const float*)d_in[4];
  const float* ln2b = (const float*)d_in[5];
  const float* Wq = (const float*)d_in[6];  const float* bq = (const float*)d_in[7];
  const float* Wk = (const float*)d_in[8];  const float* bk = (const float*)d_in[9];
  const float* Wv = (const float*)d_in[10]; const float* bv = (const float*)d_in[11];
  const float* Wo = (const float*)d_in[12]; const float* bo = (const float*)d_in[13];
  const float* W1 = (const float*)d_in[14]; const float* b1 = (const float*)d_in[15];
  const float* W2 = (const float*)d_in[16]; const float* b2 = (const float*)d_in[17];

  constexpr size_t NQ  = (size_t)Bc * SQc * Dc;    // 1,048,576
  constexpr size_t NH1 = (size_t)Bc * SQc * MLPc;  // 4,194,304
  const int M = Bc * SQc;                           // 4096 total rows

  // Carve workspace
  char* p = (char*)d_ws;
  auto takeH = [&](size_t n) { _Float16* r = (_Float16*)p; p += n * sizeof(_Float16); return r; };
  auto takeF = [&](size_t n) { float* r = (float*)p; p += n * sizeof(float); return r; };
  _Float16* xq_h   = takeH(NQ);
  _Float16* xkv_h  = takeH(NQ);
  _Float16* q_h    = takeH(NQ);               // [B,H,SQ,HD]
  _Float16* k_h    = takeH(NQ);               // [B,H,SKV,HD]
  _Float16* vt_h   = takeH(NQ);               // [B,H,HD,SKV]
  _Float16* attn_h = takeH(NQ);               // [B,SQ,D]
  _Float16* y_h    = takeH(NQ);
  _Float16* h1_h   = takeH(NH1);
  _Float16* Wq_t   = takeH((size_t)Dc * Dc);
  _Float16* Wk_t   = takeH((size_t)Dc * Dc);
  _Float16* Wv_t   = takeH((size_t)Dc * Dc);
  _Float16* Wo_t   = takeH((size_t)Dc * Dc);
  _Float16* W1_t   = takeH((size_t)Dc * MLPc);
  _Float16* W2_t   = takeH((size_t)MLPc * Dc);
  float*    x_f    = takeF(NQ);               // residual x (f32)

  float* out0 = (float*)d_out;   // x + y
  float* wout = out0 + NQ;       // attention weights [B,H,SQ,SKV]

  // LN1 on both streams
  ln_rows<<<Bc * SQc, 256, 0, stream>>>(qin, ln1g, ln1b, xq_h);
  ln_rows<<<Bc * SKVc, 256, 0, stream>>>(kvin, ln1g, ln1b, xkv_h);

  // Weight convert + transpose
  cvt_t<<<(Dc * Dc) / 256, 256, 0, stream>>>(Wq, Wq_t, Dc, Dc);
  cvt_t<<<(Dc * Dc) / 256, 256, 0, stream>>>(Wk, Wk_t, Dc, Dc);
  cvt_t<<<(Dc * Dc) / 256, 256, 0, stream>>>(Wv, Wv_t, Dc, Dc);
  cvt_t<<<(Dc * Dc) / 256, 256, 0, stream>>>(Wo, Wo_t, Dc, Dc);
  cvt_t<<<(Dc * MLPc) / 256, 256, 0, stream>>>(W1, W1_t, Dc, MLPc);
  cvt_t<<<(MLPc * Dc) / 256, 256, 0, stream>>>(W2, W2_t, MLPc, Dc);

  // Q/K projections -> head-split [B,H,S,HD]
  const long long cQ0 = (long long)Hc * SQc * HDc, cQ2 = (long long)SQc * HDc;
  gemm_wmma_kernel<<<(M / 16) * (Dc / 16) / 4, 128, 0, stream>>>(
      xq_h, Wq_t, bq, nullptr, nullptr, q_h, M, Dc, Dc, SQc, HDc, cQ0, HDc, cQ2, 1, 0);
  gemm_wmma_kernel<<<(M / 16) * (Dc / 16) / 4, 128, 0, stream>>>(
      xkv_h, Wk_t, bk, nullptr, nullptr, k_h, M, Dc, Dc, SKVc, HDc, cQ0, HDc, cQ2, 1, 0);
  // V projection -> per-head transposed [B,H,HD,SKV]
  gemm_wmma_kernel<<<(M / 16) * (Dc / 16) / 4, 128, 0, stream>>>(
      xkv_h, Wv_t, bv, nullptr, nullptr, vt_h, M, Dc, Dc, SKVc, HDc,
      (long long)Hc * HDc * SKVc, 1, (long long)HDc * SKVc, SKVc, 0);

  // Fused attention (scores+softmax+weights-out+attn*V)
  constexpr int SMEM_ATTN =
      (16 * (SKVc + 4) + 4 * 2 * 8 * 32 + 16 * 8 + 16) * (int)sizeof(float);
  attn_fused_kernel<<<Bc * Hc * (SQc / 16), 128, SMEM_ATTN, stream>>>(
      q_h, k_h, vt_h, wout, attn_h);

  // O projection + residual -> x (f32)
  gemm_wmma_kernel<<<(M / 16) * (Dc / 16) / 4, 128, 0, stream>>>(
      attn_h, Wo_t, bo, qin, x_f, nullptr, M, Dc, Dc, SQc, Dc,
      (long long)SQc * Dc, Dc, 0, 1, 0);

  // LN2 -> y_h (f16)
  ln_rows<<<M, 256, 0, stream>>>(x_f, ln2g, ln2b, y_h);

  // MLP1 with exact GELU -> h1_h (f16)
  gemm_wmma_kernel<<<(M / 16) * (MLPc / 16) / 4, 128, 0, stream>>>(
      y_h, W1_t, b1, nullptr, nullptr, h1_h, M, MLPc, Dc, SQc, MLPc,
      (long long)SQc * MLPc, MLPc, 0, 1, 1);

  // MLP2 + residual(x) -> out0 = x + y
  gemm_wmma_kernel<<<(M / 16) * (Dc / 16) / 4, 128, 0, stream>>>(
      h1_h, W2_t, b2, x_f, out0, nullptr, M, Dc, MLPc, SQc, Dc,
      (long long)SQc * Dc, Dc, 0, 1, 0);
}
